// TransformerBlock_68891275428193
// MI455X (gfx1250) — matrix operand order repaired and checked by reading, not compiled, not run on hardware
//
#include <hip/hip_runtime.h>
#include <hip/hip_bf16.h>
#include <math.h>

typedef __bf16 bf16_t;
typedef __attribute__((ext_vector_type(16))) __bf16 v16bf;
typedef __attribute__((ext_vector_type(8)))  __bf16 v8bf;
typedef __attribute__((ext_vector_type(8)))  float  v8f;
typedef __attribute__((ext_vector_type(4)))  float  v4f;
typedef __attribute__((ext_vector_type(4)))  unsigned int u32x4;
typedef __attribute__((ext_vector_type(8)))  int i32x8;
typedef __attribute__((ext_vector_type(4)))  int i32x4;

namespace cfg {
constexpr int Bb = 2, S = 2048, Dm = 1024, H = 16, HD = 64, F = 4096;
constexpr int BS = Bb * S;   // 4096 rows
}

#define WMMA_BF16(a, b, c) \
  __builtin_amdgcn_wmma_f32_16x16x32_bf16(false, (a), false, (b), (short)0, (c), false, false)

static __device__ __forceinline__ bf16_t f2bf(float f) { return (bf16_t)f; }

static __device__ __forceinline__ v16bf pack16(v8bf lo, v8bf hi) {
  v16bf r;
#pragma unroll
  for (int i = 0; i < 8; ++i) { r[i] = lo[i]; r[8 + i] = hi[i]; }
  return r;
}

// reductions across the 16-lane N-group of the WMMA C layout (wave32)
static __device__ __forceinline__ float red16_max(float v) {
#pragma unroll
  for (int m = 1; m <= 8; m <<= 1) v = fmaxf(v, __shfl_xor(v, m, 32));
  return v;
}
static __device__ __forceinline__ float red16_sum(float v) {
#pragma unroll
  for (int m = 1; m <= 8; m <<= 1) v += __shfl_xor(v, m, 32);
  return v;
}

// ---------------------------------------------------------------- TDM issue
// 2D tile load via Tensor Data Mover.  D# layout per CDNA5 ISA ch.8:
//  group0: [1:0]=count=1, [63:32]=lds byte addr, [120:64]=global addr, [127:126]=type 2
//  group1: [17:16]=data_size(1 -> 2 bytes), [79:48]=tensor_dim0, [111:80]=tensor_dim1,
//          [127:112]=tile_dim0, [143:128]=tile_dim1, [207:160]=tensor_dim0_stride
static __device__ __forceinline__ void tdm_load_tile_bf16(const void* gaddr,
                                                          unsigned int lds_off,
                                                          unsigned int tensor_d0,
                                                          unsigned int tensor_d1,
                                                          unsigned int tile_d0,
                                                          unsigned int tile_d1,
                                                          unsigned long long stride0) {
  unsigned long long ga = (unsigned long long)(uintptr_t)gaddr;
  u32x4 g0;
  g0[0] = 1u;                                   // count=1, user descriptor
  g0[1] = lds_off;
  g0[2] = (unsigned int)ga;
  g0[3] = (unsigned int)(ga >> 32) | (2u << 30); // type = 2 (image/tensor)
  i32x8 g1;
  g1[0] = (int)(1u << 16);                      // data_size = 2 bytes, no multicast
  g1[1] = (int)((tensor_d0 & 0xFFFFu) << 16);
  g1[2] = (int)((tensor_d0 >> 16) | ((tensor_d1 & 0xFFFFu) << 16));
  g1[3] = (int)((tensor_d1 >> 16) | (tile_d0 << 16));
  g1[4] = (int)(tile_d1 & 0xFFFFu);             // tile_dim2 = 0 (2D)
  g1[5] = (int)(unsigned int)(stride0 & 0xFFFFFFFFull);
  g1[6] = (int)(unsigned int)(stride0 >> 32);   // tensor_dim1_stride = 0 (2D)
  g1[7] = 0;
#if defined(__clang_major__) && __clang_major__ >= 23
  __builtin_amdgcn_tensor_load_to_lds(g0, g1, (i32x4)(0), (i32x4)(0), (i32x8)(0), 0);
#else
  __builtin_amdgcn_tensor_load_to_lds(g0, g1, (i32x4)(0), (i32x4)(0), 0);
#endif
}

static __device__ __forceinline__ unsigned int lds_offset_of(const void* p) {
  // generic pointer to __shared__ carries the LDS offset in its low 32 bits
  return (unsigned int)(uintptr_t)p;
}

// ---------------------------------------------------------------- casts
__global__ __launch_bounds__(256) void cast_f32_to_bf16(const float* __restrict__ src,
                                                        bf16_t* __restrict__ dst, int n4) {
  int id = blockIdx.x * 256 + threadIdx.x;
  if (id < n4) {
    v4f v = ((const v4f*)src)[id];
#pragma unroll
    for (int j = 0; j < 4; ++j) dst[id * 4 + j] = f2bf(v[j]);
  }
}

// ---------------------------------------------------------------- rmsnorm
__global__ __launch_bounds__(256) void rmsnorm_to_bf16(const float* __restrict__ x,
                                                       const float* __restrict__ g,
                                                       bf16_t* __restrict__ out) {
  using namespace cfg;
  const int row = blockIdx.x;
  const float* xr = x + (size_t)row * Dm;
  v4f v = ((const v4f*)xr)[threadIdx.x];
  float ss = v[0] * v[0] + v[1] * v[1] + v[2] * v[2] + v[3] * v[3];
  __shared__ float red[256];
  red[threadIdx.x] = ss;
  __syncthreads();
#pragma unroll
  for (int s = 128; s > 0; s >>= 1) {
    if (threadIdx.x < s) red[threadIdx.x] += red[threadIdx.x + s];
    __syncthreads();
  }
  float inv = rsqrtf(red[0] * (1.0f / Dm) + 1e-5f);
  v4f gv = ((const v4f*)g)[threadIdx.x];
#pragma unroll
  for (int j = 0; j < 4; ++j)
    out[(size_t)row * Dm + threadIdx.x * 4 + j] = f2bf(v[j] * inv * gv[j]);
}

// ---------------------------------------------------------------- GEMM (TDM-staged B)
// C[M,N] = A[M,K](bf16, K-contig) * W[N,K]^T (bf16, K-contig)
// Block = 8 waves, tile 64x64; weight tile 64(rows)x64(K) double-buffered in LDS via TDM.
// All fragments of a chunk are loaded into distinct registers BEFORE the WMMA chain so
// the DS/VMEM loads can be claused and overlapped with the matrix pipe (partial waits).
// MODE 0: bf16 out.  MODE 1: fp32 out + fp32 residual.
template <int MODE>
__global__ __launch_bounds__(256, 2) void gemm_bf16(const bf16_t* __restrict__ A,
                                                    const bf16_t* __restrict__ W,
                                                    const float* __restrict__ res,
                                                    bf16_t* __restrict__ outB,
                                                    float* __restrict__ outF, int N, int K) {
  __shared__ bf16_t Btile[2][64 * 64];

  const int lane = threadIdx.x & 31;
  const int wave = threadIdx.x >> 5;
  const int half = lane >> 4;
  const int l16 = lane & 15;
  const int m0 = blockIdx.y * 64 + (wave & 3) * 16;
  const int nblk = blockIdx.x * 64;
  const int nloc0 = (wave >> 2) * 32 + l16;    // local B row of first N-frag
  const int nloc1 = nloc0 + 16;
  const bool w0 = (threadIdx.x < 32);          // wave 0 drives the TDM

  const bf16_t* Ap = A + (size_t)(m0 + l16) * K + half * 16;  // A-frag layout

  const int nchunks = K / 64;
  auto issue = [&](int chunk, int buf) {
    tdm_load_tile_bf16(W + (size_t)nblk * K + (size_t)chunk * 64,
                       lds_offset_of(&Btile[buf][0]),
                       (unsigned)K, 64u, 64u, 64u, (unsigned long long)K);
  };
  if (w0) {
    issue(0, 0);
    if (nchunks > 1) issue(1, 1);
  }

  v8f acc0 = {}; v8f acc1 = {};
  for (int c = 0; c < nchunks; ++c) {
    if (w0) {
      if (c + 1 < nchunks) __builtin_amdgcn_s_wait_tensorcnt(1);
      else                 __builtin_amdgcn_s_wait_tensorcnt(0);
    }
    __syncthreads();
    const bf16_t* bt = Btile[c & 1];
    const int k0 = c * 64;

    // gather ALL fragments for this 64-wide K chunk first (distinct registers)
    v16bf a0 = pack16(*(const v8bf*)(Ap + k0),      *(const v8bf*)(Ap + k0 + 8));
    v16bf a1 = pack16(*(const v8bf*)(Ap + k0 + 32), *(const v8bf*)(Ap + k0 + 40));
    const bf16_t* p00 = bt + nloc0 * 64 + half * 16;        // ks=0, frag0
    const bf16_t* p10 = bt + nloc1 * 64 + half * 16;        // ks=0, frag1
    const bf16_t* p01 = p00 + 32;                           // ks=1, frag0
    const bf16_t* p11 = p10 + 32;                           // ks=1, frag1
    v16bf b00 = pack16(*(const v8bf*)p00, *(const v8bf*)(p00 + 8));
    v16bf b10 = pack16(*(const v8bf*)p10, *(const v8bf*)(p10 + 8));
    v16bf b01 = pack16(*(const v8bf*)p01, *(const v8bf*)(p01 + 8));
    v16bf b11 = pack16(*(const v8bf*)p11, *(const v8bf*)(p11 + 8));

    acc0 = WMMA_BF16(a0, b00, acc0);
    acc1 = WMMA_BF16(a0, b10, acc1);
    acc0 = WMMA_BF16(a1, b01, acc0);
    acc1 = WMMA_BF16(a1, b11, acc1);

    __builtin_prefetch(Ap + k0 + 512, 0, 3);   // near-scope A prefetch
    __syncthreads();
    if (w0 && c + 2 < nchunks) issue(c + 2, c & 1);
  }

#pragma unroll
  for (int r = 0; r < 8; ++r) {
    size_t row = (size_t)(m0 + r + half * 8);
    size_t c0 = (size_t)(nblk + nloc0);
    size_t c1 = (size_t)(nblk + nloc1);
    if (MODE == 0) {
      outB[row * N + c0] = f2bf(acc0[r]);
      outB[row * N + c1] = f2bf(acc1[r]);
    } else {
      outF[row * N + c0] = acc0[r] + res[row * N + c0];
      outF[row * N + c1] = acc1[r] + res[row * N + c1];
    }
  }
}

// ---------------------------------------------------------------- SwiGLU dual GEMM
// u = silu(A*W1^T) * (A*W3^T), bf16 out; W1 and W3 tiles both staged via TDM.
__global__ __launch_bounds__(256, 2) void gemm_swiglu_bf16(const bf16_t* __restrict__ A,
                                                           const bf16_t* __restrict__ W1,
                                                           const bf16_t* __restrict__ W3,
                                                           bf16_t* __restrict__ U, int N, int K) {
  __shared__ bf16_t Bt1[2][64 * 64];
  __shared__ bf16_t Bt3[2][64 * 64];

  const int lane = threadIdx.x & 31;
  const int wave = threadIdx.x >> 5;
  const int half = lane >> 4;
  const int l16 = lane & 15;
  const int m0 = blockIdx.y * 64 + (wave & 3) * 16;
  const int nblk = blockIdx.x * 64;
  const int nloc0 = (wave >> 2) * 32 + l16;
  const int nloc1 = nloc0 + 16;
  const bool w0 = (threadIdx.x < 32);

  const bf16_t* Ap = A + (size_t)(m0 + l16) * K + half * 16;

  const int nchunks = K / 64;
  auto issue = [&](int chunk, int buf) {
    tdm_load_tile_bf16(W1 + (size_t)nblk * K + (size_t)chunk * 64,
                       lds_offset_of(&Bt1[buf][0]),
                       (unsigned)K, 64u, 64u, 64u, (unsigned long long)K);
    tdm_load_tile_bf16(W3 + (size_t)nblk * K + (size_t)chunk * 64,
                       lds_offset_of(&Bt3[buf][0]),
                       (unsigned)K, 64u, 64u, 64u, (unsigned long long)K);
  };
  if (w0) {
    issue(0, 0);
    if (nchunks > 1) issue(1, 1);
  }

  v8f p10 = {}, p11 = {}, p30 = {}, p31 = {};
  for (int c = 0; c < nchunks; ++c) {
    if (w0) {
      if (c + 1 < nchunks) __builtin_amdgcn_s_wait_tensorcnt(2);
      else                 __builtin_amdgcn_s_wait_tensorcnt(0);
    }
    __syncthreads();
    const bf16_t* bt1 = Bt1[c & 1];
    const bf16_t* bt3 = Bt3[c & 1];
    const int k0 = c * 64;
    const int off0 = nloc0 * 64 + half * 16;
    const int off1 = nloc1 * 64 + half * 16;

    v16bf a0 = pack16(*(const v8bf*)(Ap + k0),      *(const v8bf*)(Ap + k0 + 8));
    v16bf a1 = pack16(*(const v8bf*)(Ap + k0 + 32), *(const v8bf*)(Ap + k0 + 40));
    v16bf c100 = pack16(*(const v8bf*)(bt1 + off0),      *(const v8bf*)(bt1 + off0 + 8));
    v16bf c110 = pack16(*(const v8bf*)(bt1 + off1),      *(const v8bf*)(bt1 + off1 + 8));
    v16bf c101 = pack16(*(const v8bf*)(bt1 + off0 + 32), *(const v8bf*)(bt1 + off0 + 40));
    v16bf c111 = pack16(*(const v8bf*)(bt1 + off1 + 32), *(const v8bf*)(bt1 + off1 + 40));
    v16bf c300 = pack16(*(const v8bf*)(bt3 + off0),      *(const v8bf*)(bt3 + off0 + 8));
    v16bf c310 = pack16(*(const v8bf*)(bt3 + off1),      *(const v8bf*)(bt3 + off1 + 8));
    v16bf c301 = pack16(*(const v8bf*)(bt3 + off0 + 32), *(const v8bf*)(bt3 + off0 + 40));
    v16bf c311 = pack16(*(const v8bf*)(bt3 + off1 + 32), *(const v8bf*)(bt3 + off1 + 40));

    p10 = WMMA_BF16(a0, c100, p10);
    p11 = WMMA_BF16(a0, c110, p11);
    p30 = WMMA_BF16(a0, c300, p30);
    p31 = WMMA_BF16(a0, c310, p31);
    p10 = WMMA_BF16(a1, c101, p10);
    p11 = WMMA_BF16(a1, c111, p11);
    p30 = WMMA_BF16(a1, c301, p30);
    p31 = WMMA_BF16(a1, c311, p31);

    __builtin_prefetch(Ap + k0 + 512, 0, 3);
    __syncthreads();
    if (w0 && c + 2 < nchunks) issue(c + 2, c & 1);
  }

#pragma unroll
  for (int r = 0; r < 8; ++r) {
    size_t row = (size_t)(m0 + r + half * 8);
    float s0 = p10[r] / (1.0f + __expf(-p10[r]));  // silu
    float s1 = p11[r] / (1.0f + __expf(-p11[r]));
    U[row * N + (nblk + nloc0)] = f2bf(s0 * p30[r]);
    U[row * N + (nblk + nloc1)] = f2bf(s1 * p31[r]);
  }
}

// ---------------------------------------------------------------- flash attention
// One wave per (b, h, 16-row q tile).  Causal, scale = 1/sqrt(64).
__global__ __launch_bounds__(32) void attn_fwd(const bf16_t* __restrict__ Q,
                                               const bf16_t* __restrict__ Km,
                                               const bf16_t* __restrict__ V,
                                               bf16_t* __restrict__ O) {
  using namespace cfg;
  const int qt = blockIdx.x & (S / 16 - 1);        // 128 tiles -> low 7 bits
  const int hh = (blockIdx.x >> 7) & (H - 1);
  const int b  = blockIdx.x >> 11;
  const int q0 = qt * 16;
  const int lane = threadIdx.x;
  const int half = lane >> 4;
  const int l16 = lane & 15;

  __shared__ bf16_t Pb[16 * 32];   // probs tile, C-layout -> A-layout relayout
  __shared__ bf16_t Vt[64 * 48];   // V^T tile: [d][key], padded row stride 48

  const size_t base = (size_t)b * S;

  v16bf qa[2];
  {
    const bf16_t* qp = Q + (base + q0 + l16) * Dm + hh * HD + half * 16;
#pragma unroll
    for (int ks = 0; ks < 2; ++ks)
      qa[ks] = pack16(*(const v8bf*)(qp + ks * 32), *(const v8bf*)(qp + ks * 32 + 8));
  }

  v8f Oacc[4] = {};
  float mrow[8], lrow[8];
#pragma unroll
  for (int r = 0; r < 8; ++r) { mrow[r] = -1e30f; lrow[r] = 0.0f; }

  const int nkb = (q0 + 16 + 31) / 32;
  for (int kb = 0; kb < nkb; ++kb) {
    const int k0 = kb * 32;
    v16bf kf[2][2];
#pragma unroll
    for (int nt = 0; nt < 2; ++nt) {
      const bf16_t* kp = Km + (base + k0 + nt * 16 + l16) * Dm + hh * HD + half * 16;
#pragma unroll
      for (int ks = 0; ks < 2; ++ks)
        kf[nt][ks] = pack16(*(const v8bf*)(kp + ks * 32), *(const v8bf*)(kp + ks * 32 + 8));
    }
    v8f s0 = {}, s1 = {};
    s0 = WMMA_BF16(qa[0], kf[0][0], s0);
    s0 = WMMA_BF16(qa[1], kf[0][1], s0);
    s1 = WMMA_BF16(qa[0], kf[1][0], s1);
    s1 = WMMA_BF16(qa[1], kf[1][1], s1);

#pragma unroll
    for (int r = 0; r < 8; ++r) {
      const int qrow = q0 + r + half * 8;
      float v0 = (k0 + l16      <= qrow) ? s0[r] * 0.125f : -1e30f;
      float v1 = (k0 + 16 + l16 <= qrow) ? s1[r] * 0.125f : -1e30f;
      float rm = red16_max(fmaxf(v0, v1));
      float mn = fmaxf(mrow[r], rm);
      float corr = __expf(mrow[r] - mn);
      float p0 = __expf(v0 - mn);
      float p1 = __expf(v1 - mn);
      lrow[r] = lrow[r] * corr + red16_sum(p0 + p1);
      mrow[r] = mn;
#pragma unroll
      for (int g = 0; g < 4; ++g) Oacc[g][r] *= corr;
      Pb[(r + half * 8) * 32 + l16]      = f2bf(p0);
      Pb[(r + half * 8) * 32 + 16 + l16] = f2bf(p1);
    }
    __builtin_amdgcn_wave_barrier();   // per-wave DS ops are in-order; fence compiler

    {
      const bf16_t* vp = V + (base + k0 + lane) * Dm + hh * HD;
#pragma unroll
      for (int c = 0; c < 8; ++c) {
        v8bf ch = *(const v8bf*)(vp + c * 8);
#pragma unroll
        for (int j = 0; j < 8; ++j) Vt[(c * 8 + j) * 48 + lane] = ch[j];
      }
    }
    __builtin_amdgcn_wave_barrier();

    v16bf pf;
    {
      const int prow = l16;
      const int cb = half * 16;
#pragma unroll
      for (int j = 0; j < 8; ++j) {
        pf[j]     = Pb[prow * 32 + cb + j];
        pf[8 + j] = Pb[prow * 32 + cb + 8 + j];
      }
    }
#pragma unroll
    for (int g = 0; g < 4; ++g) {
      const bf16_t* vt = &Vt[(g * 16 + l16) * 48 + half * 16];
      v16bf vf = pack16(*(const v8bf*)vt, *(const v8bf*)(vt + 8));
      Oacc[g] = WMMA_BF16(pf, vf, Oacc[g]);
    }
    __builtin_amdgcn_wave_barrier();
  }

#pragma unroll
  for (int g = 0; g < 4; ++g)
#pragma unroll
    for (int r = 0; r < 8; ++r) {
      const size_t row = base + q0 + r + half * 8;
      O[row * Dm + hh * HD + g * 16 + l16] = f2bf(Oacc[g][r] / lrow[r]);
    }
}

// ---------------------------------------------------------------- launcher
extern "C" void kernel_launch(void* const* d_in, const int* in_sizes, int n_in,
                              void* d_out, int out_size, void* d_ws, size_t ws_size,
                              hipStream_t stream) {
  using namespace cfg;
  (void)in_sizes; (void)n_in; (void)out_size; (void)ws_size;

  const float* x  = (const float*)d_in[0];
  const float* g1 = (const float*)d_in[1];
  const float* wq = (const float*)d_in[2];
  const float* wk = (const float*)d_in[3];
  const float* wv = (const float*)d_in[4];
  const float* wo = (const float*)d_in[5];
  const float* g2 = (const float*)d_in[6];
  const float* w1 = (const float*)d_in[7];
  const float* w2 = (const float*)d_in[8];
  const float* w3 = (const float*)d_in[9];

  char* p = (char*)d_ws;
  auto take = [&](size_t bytes) -> char* {
    char* r = p;
    p += (bytes + 255) & ~(size_t)255;
    return r;
  };
  bf16_t* wq_b = (bf16_t*)take((size_t)Dm * Dm * 2);
  bf16_t* wk_b = (bf16_t*)take((size_t)Dm * Dm * 2);
  bf16_t* wv_b = (bf16_t*)take((size_t)Dm * Dm * 2);
  bf16_t* wo_b = (bf16_t*)take((size_t)Dm * Dm * 2);
  bf16_t* w1_b = (bf16_t*)take((size_t)F * Dm * 2);
  bf16_t* w2_b = (bf16_t*)take((size_t)Dm * F * 2);
  bf16_t* w3_b = (bf16_t*)take((size_t)F * Dm * 2);
  bf16_t* xn_b = (bf16_t*)take((size_t)BS * Dm * 2);
  bf16_t* q_b  = (bf16_t*)take((size_t)BS * Dm * 2);
  bf16_t* k_b  = (bf16_t*)take((size_t)BS * Dm * 2);
  bf16_t* v_b  = (bf16_t*)take((size_t)BS * Dm * 2);
  bf16_t* o_b  = (bf16_t*)take((size_t)BS * Dm * 2);
  float*  hbuf = (float*) take((size_t)BS * Dm * 4);
  bf16_t* hn_b = (bf16_t*)take((size_t)BS * Dm * 2);
  bf16_t* u_b  = (bf16_t*)take((size_t)BS * F * 2);

  auto castN = [&](const float* s, bf16_t* d, size_t n) {
    int n4 = (int)(n / 4);
    cast_f32_to_bf16<<<(n4 + 255) / 256, 256, 0, stream>>>(s, d, n4);
  };
  castN(wq, wq_b, (size_t)Dm * Dm);
  castN(wk, wk_b, (size_t)Dm * Dm);
  castN(wv, wv_b, (size_t)Dm * Dm);
  castN(wo, wo_b, (size_t)Dm * Dm);
  castN(w1, w1_b, (size_t)F * Dm);
  castN(w2, w2_b, (size_t)Dm * F);
  castN(w3, w3_b, (size_t)F * Dm);

  // attention sublayer
  rmsnorm_to_bf16<<<BS, 256, 0, stream>>>(x, g1, xn_b);

  dim3 blk(256);
  dim3 gD(Dm / 64, BS / 64);
  gemm_bf16<0><<<gD, blk, 0, stream>>>(xn_b, wq_b, nullptr, q_b, nullptr, Dm, Dm);
  gemm_bf16<0><<<gD, blk, 0, stream>>>(xn_b, wk_b, nullptr, k_b, nullptr, Dm, Dm);
  gemm_bf16<0><<<gD, blk, 0, stream>>>(xn_b, wv_b, nullptr, v_b, nullptr, Dm, Dm);

  attn_fwd<<<Bb * H * (S / 16), 32, 0, stream>>>(q_b, k_b, v_b, o_b);

  // h = x + o * wo^T   (fp32)
  gemm_bf16<1><<<gD, blk, 0, stream>>>(o_b, wo_b, x, nullptr, hbuf, Dm, Dm);

  // SwiGLU sublayer
  rmsnorm_to_bf16<<<BS, 256, 0, stream>>>(hbuf, g2, hn_b);

  dim3 gF(F / 64, BS / 64);
  gemm_swiglu_bf16<<<gF, blk, 0, stream>>>(hn_b, w1_b, w3_b, u_b, F, Dm);

  // out = h + u * w2^T  (fp32, K = F)
  gemm_bf16<1><<<gD, blk, 0, stream>>>(u_b, w2_b, hbuf, nullptr, (float*)d_out, Dm, F);
}
